// SelfAttention_22926535426074
// MI455X (gfx1250) — matrix operand order repaired and checked by reading, not compiled, not run on hardware
//
#include <hip/hip_runtime.h>
#include <hip/hip_bf16.h>

typedef __bf16 bf16_t;
typedef __attribute__((ext_vector_type(16))) __bf16    v16bf;
typedef __attribute__((ext_vector_type(8)))  __bf16    v8bf;
typedef __attribute__((ext_vector_type(8)))  float     v8f;
typedef __attribute__((ext_vector_type(4)))  unsigned  v4u;
typedef __attribute__((ext_vector_type(8)))  unsigned  v8u;

// ---------------------------------------------------------------------------
// WMMA helpers (CDNA5 wave32, v_wmma_f32_16x16x32_bf16)
// ---------------------------------------------------------------------------
__device__ __forceinline__ v8f wmma_bf16(v16bf a, v16bf b, v8f c) {
  return __builtin_amdgcn_wmma_f32_16x16x32_bf16(
      false, a, false, b, (short)0, c, false, false);
}

// A-frag: 16x32 (MxK) bf16, row-major source with row stride lda (elements).
// lanes 0-15 row M=lane, K={0..7,16..23}; lanes 16-31 row M=lane-16, K={8..15,24..31}
__device__ __forceinline__ v16bf load_a_frag(const bf16_t* base, int lda) {
  int lane = threadIdx.x & 31;
  int m    = lane & 15;
  int koff = (lane < 16) ? 0 : 8;
  const bf16_t* p = base + (long)m * lda + koff;
  union { v16bf v; v8bf h[2]; } u;
  u.h[0] = *(const v8bf*)(p);
  u.h[1] = *(const v8bf*)(p + 16);
  return u.v;
}

// B-frag: 32x16 (KxN) bf16 where B(k,n) = base[n*ldb + k].
// lanes 0-15: col N=lane, K={0..7,16..23}; lanes 16-31: col N=lane-16, K={8..15,24..31}
__device__ __forceinline__ v16bf load_b_frag(const bf16_t* base, int ldb) {
  int lane = threadIdx.x & 31;
  int n    = lane & 15;
  int koff = (lane < 16) ? 0 : 8;
  const bf16_t* p = base + (long)n * ldb + koff;
  union { v16bf v; v8bf h[2]; } u;
  u.h[0] = *(const v8bf*)(p);
  u.h[1] = *(const v8bf*)(p + 16);
  return u.v;
}

// ---------------------------------------------------------------------------
// Kernel 1: fp32 LayerNorm over D=4096, one block (256 thr) per row
// ---------------------------------------------------------------------------
__global__ __launch_bounds__(256) void layernorm_kernel(
    const bf16_t* __restrict__ x, const float* __restrict__ w,
    bf16_t* __restrict__ y, int D) {
  long row = blockIdx.x;
  const bf16_t* xr = x + row * D;
  bf16_t*       yr = y + row * D;
  float vals[16];
  float lsum = 0.f, lsq = 0.f;
#pragma unroll
  for (int i = 0; i < 16; ++i) {
    float v = (float)xr[threadIdx.x + 256 * i];
    vals[i] = v; lsum += v; lsq += v * v;
  }
#pragma unroll
  for (int off = 16; off; off >>= 1) {
    lsum += __shfl_xor(lsum, off, 32);
    lsq  += __shfl_xor(lsq,  off, 32);
  }
  __shared__ float red0[8], red1[8];
  int wave = threadIdx.x >> 5, lane = threadIdx.x & 31;
  if (lane == 0) { red0[wave] = lsum; red1[wave] = lsq; }
  __syncthreads();
  if (threadIdx.x == 0) {
    float s = 0.f, q = 0.f;
#pragma unroll
    for (int i = 0; i < 8; ++i) { s += red0[i]; q += red1[i]; }
    red0[0] = s; red1[0] = q;
  }
  __syncthreads();
  float mean = red0[0] / (float)D;
  float var  = red1[0] / (float)D - mean * mean;
  float rstd = rsqrtf(var + 1e-5f);
#pragma unroll
  for (int i = 0; i < 16; ++i) {
    int c = threadIdx.x + 256 * i;
    yr[c] = (bf16_t)((vals[i] - mean) * rstd * w[c]);
  }
}

// ---------------------------------------------------------------------------
// Kernel 2: C[M,N] = A[M,K] * W[N,K]^T   (bf16 in, f32 acc, bf16 out)
// 256 thr = 8 waves in a 4(M) x 2(N) grid; wave computes 32x64 (8 WMMA per
// k-step of 32, 6 fragment loads). Next k-step's fragments are loaded before
// the current WMMAs issue (register double-buffering) so v_wmma never waits
// on its own operands' loads. Block tile 128x128.
// ---------------------------------------------------------------------------
__global__ __launch_bounds__(256) void gemm_bf16_nt(
    const bf16_t* __restrict__ A, const bf16_t* __restrict__ W,
    bf16_t* __restrict__ C, int M, int N, int K) {
  int wave = threadIdx.x >> 5;
  int wm = wave & 3, wn = wave >> 2;
  int m0 = blockIdx.x * 128 + wm * 32;
  int n0 = blockIdx.y * 128 + wn * 64;

  const bf16_t* A0 = A + (long)m0 * K;
  const bf16_t* A1 = A + (long)(m0 + 16) * K;
  const bf16_t* Wt[4];
#pragma unroll
  for (int t = 0; t < 4; ++t) Wt[t] = W + (long)(n0 + 16 * t) * K;

  v8f acc[2][4] = {};
  // prologue: fragments for k0 = 0
  v16bf a0 = load_a_frag(A0, K);
  v16bf a1 = load_a_frag(A1, K);
  v16bf b[4];
#pragma unroll
  for (int t = 0; t < 4; ++t) b[t] = load_b_frag(Wt[t], K);

  for (int k0 = 0; k0 < K; k0 += 32) {
    int k1 = (k0 + 32 < K) ? (k0 + 32) : k0;   // last iter: benign reload
    // issue next k-step loads first so they overlap the WMMAs below
    v16bf a0n = load_a_frag(A0 + k1, K);
    v16bf a1n = load_a_frag(A1 + k1, K);
    v16bf bn[4];
#pragma unroll
    for (int t = 0; t < 4; ++t) bn[t] = load_b_frag(Wt[t] + k1, K);
#pragma unroll
    for (int t = 0; t < 4; ++t) acc[0][t] = wmma_bf16(a0, b[t], acc[0][t]);
#pragma unroll
    for (int t = 0; t < 4; ++t) acc[1][t] = wmma_bf16(a1, b[t], acc[1][t]);
    a0 = a0n; a1 = a1n;
#pragma unroll
    for (int t = 0; t < 4; ++t) b[t] = bn[t];
  }

  int lane = threadIdx.x & 31;
  int n  = lane & 15;
  int mb = (lane < 16) ? 0 : 8;
#pragma unroll
  for (int i = 0; i < 2; ++i)
#pragma unroll
    for (int t = 0; t < 4; ++t)
#pragma unroll
      for (int r = 0; r < 8; ++r)
        C[(long)(m0 + 16 * i + mb + r) * N + (n0 + 16 * t + n)] =
            (bf16_t)acc[i][t][r];
}

// ---------------------------------------------------------------------------
// Kernel 3: interleaved RoPE (torchtune convention), in place, fp32 math
// ---------------------------------------------------------------------------
__global__ void rope_kernel(bf16_t* __restrict__ t, int S, int H, long total) {
  long idx = (long)blockIdx.x * blockDim.x + threadIdx.x;
  if (idx >= total) return;
  int  i  = (int)(idx & 63);
  long rh = idx >> 6;
  int  s  = (int)((rh / H) % S);
  float freq = __expf(-(2.0f * (float)i / 128.0f) * __logf(10000.0f));
  float ang  = (float)s * freq;
  float sn, cs;
  __sincosf(ang, &sn, &cs);
  bf16_t* p = t + rh * 128 + 2 * i;
  float t0 = (float)p[0], t1 = (float)p[1];
  p[0] = (bf16_t)(t0 * cs - t1 * sn);
  p[1] = (bf16_t)(t1 * cs + t0 * sn);
}

// ---------------------------------------------------------------------------
// Kernel 4: causal GQA flash attention.
// grid (S/64, HQ, B), block 128 thr (4 waves); wave owns 16 query rows.
// K tile (32 keys x 128 dims, 2B elems) staged global->LDS by the Tensor
// Data Mover (tensor_load_to_lds, TENSORcnt); V staged transposed by VALU.
// ---------------------------------------------------------------------------
__global__ __launch_bounds__(128) void flash_attn_kernel(
    const bf16_t* __restrict__ Q, const bf16_t* __restrict__ Kb,
    const bf16_t* __restrict__ Vb, bf16_t* __restrict__ O,
    int S, int HQ_, int HKV_, float scale) {
  const int HD_ = 128;
  int qtile = blockIdx.x, hq = blockIdx.y, b = blockIdx.z;
  int G_   = HQ_ / HKV_;
  int hkv  = hq / G_;
  int wave = threadIdx.x >> 5;
  int lane = threadIdx.x & 31;
  int q0   = qtile * 64 + wave * 16;

  __shared__ __align__(16) bf16_t Klds[32 * 128];   // [key][d] (TDM dest)
  __shared__ __align__(16) bf16_t Vt[128 * 32];     // [d][key] transposed
  __shared__ __align__(16) bf16_t Pbuf[4][16 * 32];

  const int ldq = HQ_ * HD_;
  const int ldk = HKV_ * HD_;
  const bf16_t* qbase  = Q  + (((long)b * S + q0) * HQ_ + hq) * HD_;
  const bf16_t* kbhead = Kb + ((long)b * S * HKV_ + hkv) * HD_;
  const bf16_t* vbhead = Vb + ((long)b * S * HKV_ + hkv) * HD_;

  v16bf qf[4];
#pragma unroll
  for (int dc = 0; dc < 4; ++dc) qf[dc] = load_a_frag(qbase + dc * 32, ldq);

  v8f o[8] = {};
  float rmax[8], rsum[8];
#pragma unroll
  for (int r = 0; r < 8; ++r) { rmax[r] = -1e30f; rsum[r] = 0.f; }

  int n  = lane & 15;
  int mb = (lane < 16) ? 0 : 8;
  int numTiles = qtile * 2 + 2;                    // uniform causal bound

  // TDM descriptor pieces that do not change across key tiles
  unsigned ldsAddr = (unsigned)(unsigned long)(void*)&Klds[0];

  for (int kt = 0; kt < numTiles; ++kt) {
    int kbase = kt * 32;

    // --- TDM: K tile (32 rows x 256B, row stride ldk*2B) global -> LDS ---
    if (threadIdx.x < 32) {
      unsigned long ga = (unsigned long)(const void*)(kbhead + (long)kbase * ldk);
      // D# group 0: count=1 | lds_addr | 57-bit global addr | type=2
      v4u g0 = { 1u, ldsAddr, (unsigned)ga,
                 (unsigned)(ga >> 32) | 0x80000000u };
      // D# group 1: data_size=2B; tensor_dim0=128 (d), tensor_dim1=S (keys);
      // tile 128 x 32; tensor_dim0_stride = HKV*HD elements
      unsigned dim0 = 128u, dim1 = (unsigned)S;
      unsigned tile0 = 128u, tile1 = 32u;
      unsigned str0 = (unsigned)ldk;
      v8u g1 = { 0x00010000u,                      // data_size=1 -> 2 bytes
                 (dim0 & 0xFFFFu) << 16,           // tensor_dim0[15:0]
                 (dim0 >> 16) | ((dim1 & 0xFFFFu) << 16),
                 (dim1 >> 16) | (tile0 << 16),     // tile_dim0
                 tile1,                            // tile_dim1 (tile_dim2=0)
                 str0,                             // tensor_dim0_stride[31:0]
                 0u, 0u };                         // stride hi, dim1_stride=0
      asm volatile("tensor_load_to_lds %0, %1"
                   :: "s"(g0), "s"(g1)
                   : "memory");
      __builtin_amdgcn_s_wait_tensorcnt(0);
    }
    // --- V tile staged transposed: Vt[d][key] ---
    {
      int key = threadIdx.x >> 2;
      int d0  = (threadIdx.x & 3) * 32;
      const bf16_t* gv = vbhead + (long)(kbase + key) * ldk + d0;
#pragma unroll
      for (int j = 0; j < 32; ++j) Vt[(d0 + j) * 32 + key] = gv[j];
    }
    __syncthreads();

    bool active = (kbase <= q0 + 15);
    if (active) {
      v8f sc0 = {}, sc1 = {};
#pragma unroll
      for (int dc = 0; dc < 4; ++dc) {
        v16bf kf0 = load_b_frag(&Klds[0  * 128 + dc * 32], 128);
        v16bf kf1 = load_b_frag(&Klds[16 * 128 + dc * 32], 128);
        sc0 = wmma_bf16(qf[dc], kf0, sc0);
        sc1 = wmma_bf16(qf[dc], kf1, sc1);
      }
#pragma unroll
      for (int r = 0; r < 8; ++r) {
        int qrow = q0 + mb + r;
        float s0 = sc0[r] * scale;
        float s1 = sc1[r] * scale;
        if (kbase + n      > qrow) s0 = -1e30f;
        if (kbase + 16 + n > qrow) s1 = -1e30f;
        float mx = fmaxf(s0, s1);
#pragma unroll
        for (int off = 8; off; off >>= 1) mx = fmaxf(mx, __shfl_xor(mx, off, 32));
        float mnew = fmaxf(rmax[r], mx);
        float corr = __expf(rmax[r] - mnew);
        float p0 = __expf(s0 - mnew);
        float p1 = __expf(s1 - mnew);
        float ps = p0 + p1;
#pragma unroll
        for (int off = 8; off; off >>= 1) ps += __shfl_xor(ps, off, 32);
        rsum[r] = rsum[r] * corr + ps;
        rmax[r] = mnew;
#pragma unroll
        for (int dc = 0; dc < 8; ++dc) o[dc][r] *= corr;
        Pbuf[wave][(mb + r) * 32 + n]      = (bf16_t)p0;
        Pbuf[wave][(mb + r) * 32 + 16 + n] = (bf16_t)p1;
      }
    }
    __syncthreads();
    if (active) {
      v16bf pf = load_a_frag(&Pbuf[wave][0], 32);
#pragma unroll
      for (int dc = 0; dc < 8; ++dc) {
        v16bf vf = load_b_frag(&Vt[(dc * 16) * 32], 32);
        o[dc] = wmma_bf16(pf, vf, o[dc]);
      }
    }
    __syncthreads();
  }

  bf16_t* obase = O + (((long)b * S + q0) * HQ_ + hq) * HD_;
#pragma unroll
  for (int dc = 0; dc < 8; ++dc)
#pragma unroll
    for (int r = 0; r < 8; ++r)
      obase[(long)(mb + r) * ldq + dc * 16 + n] = (bf16_t)(o[dc][r] / rsum[r]);
}

// ---------------------------------------------------------------------------
// Host-side launch sequence
// ---------------------------------------------------------------------------
extern "C" void kernel_launch(void* const* d_in, const int* in_sizes, int n_in,
                              void* d_out, int out_size, void* d_ws, size_t ws_size,
                              hipStream_t stream) {
  (void)in_sizes; (void)n_in; (void)out_size; (void)ws_size;
  const int Bn = 2, S = 2048, D = 4096, HQn = 32, HKVn = 8, HDn = 128;
  const long M = (long)Bn * S;

  const bf16_t* x    = (const bf16_t*)d_in[0];
  const float*  ln_w = (const float*) d_in[1];
  const bf16_t* wq   = (const bf16_t*)d_in[2];
  const bf16_t* wk   = (const bf16_t*)d_in[3];
  const bf16_t* wv   = (const bf16_t*)d_in[4];
  const bf16_t* wo   = (const bf16_t*)d_in[5];
  bf16_t*       out  = (bf16_t*)d_out;

  bf16_t* xn = (bf16_t*)d_ws;                  // [M, D]
  bf16_t* q  = xn + M * D;                     // [M, HQ*HD]
  bf16_t* k  = q  + M * (HQn * HDn);           // [M, HKV*HD]
  bf16_t* v  = k  + M * (HKVn * HDn);          // [M, HKV*HD]
  bf16_t* ao = xn;                             // attn out reuses xn

  layernorm_kernel<<<(int)M, 256, 0, stream>>>(x, ln_w, xn, D);

  gemm_bf16_nt<<<dim3((int)(M / 128), (HQn * HDn) / 128), 256, 0, stream>>>(
      xn, wq, q, (int)M, HQn * HDn, D);
  gemm_bf16_nt<<<dim3((int)(M / 128), (HKVn * HDn) / 128), 256, 0, stream>>>(
      xn, wk, k, (int)M, HKVn * HDn, D);
  gemm_bf16_nt<<<dim3((int)(M / 128), (HKVn * HDn) / 128), 256, 0, stream>>>(
      xn, wv, v, (int)M, HKVn * HDn, D);

  {
    long pq = M * HQn * 64;
    long pk = M * HKVn * 64;
    rope_kernel<<<(int)((pq + 255) / 256), 256, 0, stream>>>(q, S, HQn, pq);
    rope_kernel<<<(int)((pk + 255) / 256), 256, 0, stream>>>(k, S, HKVn, pk);
  }

  {
    float scale = 0.088388347648318447f;       // 1/sqrt(128)
    dim3 grid(S / 64, HQn, Bn);
    flash_attn_kernel<<<grid, 128, 0, stream>>>(q, k, v, ao, S, HQn, HKVn, scale);
  }

  gemm_bf16_nt<<<dim3((int)(M / 128), D / 128), 256, 0, stream>>>(
      ao, wo, out, (int)M, D, D);
}
